// SA_Block_29480655520152
// MI455X (gfx1250) — matrix operand order repaired and checked by reading, not compiled, not run on hardware
//
#include <hip/hip_runtime.h>

// ---------------- types / WMMA helpers ----------------
typedef _Float16 half8  __attribute__((ext_vector_type(8)));
typedef _Float16 half16 __attribute__((ext_vector_type(16)));
typedef float    f32x8  __attribute__((ext_vector_type(8)));

union H16u { half16 v; half8 h[2]; };

__device__ __forceinline__ f32x8 wmma16(half16 a, half16 b, f32x8 c) {
    return __builtin_amdgcn_wmma_f32_16x16x32_f16(false, a, false, b, (short)0, c, false, false);
}

// A-matrix 16x32 f16, row-major storage (ld = K stride).
// lane: M = lane&15, g = lane>>4; elems 0..7 -> K = k0+g*8.., elems 8..15 -> K = k0+16+g*8..
__device__ __forceinline__ half16 load_a_frag(const _Float16* base, int row0, int ld, int k0, int lane) {
    int m = lane & 15, g = lane >> 4;
    const _Float16* p = base + (size_t)(row0 + m) * ld + k0 + g * 8;
    H16u u;
    u.h[0] = *(const half8*)(p);
    u.h[1] = *(const half8*)(p + 16);
    return u.v;
}

// B-matrix 32x16 f16, column-major storage (column contiguous over K, ld = K stride).
// lane: N = lane&15, g = lane>>4; elems 0..7 -> K = k0+g*8.., elems 8..15 -> K = k0+16+g*8..
__device__ __forceinline__ half16 load_b_frag(const _Float16* base, int col0, int ld, int k0, int lane) {
    int n = lane & 15, g = lane >> 4;
    const _Float16* p = base + (size_t)(col0 + n) * ld + k0 + g * 8;
    H16u u;
    u.h[0] = *(const half8*)(p);
    u.h[1] = *(const half8*)(p + 16);
    return u.v;
}

// ---------------- problem dims ----------------
// B=8, C=128, C8=16, D=16, H=W=32, HW=1024, DHW=16384, F=C8*D=256, M2=C*D=2048

// ---------------- stage 1: k/q 1x1x1 conv -> f16 staged [b][hw][f=c8*16+d] ----------------
__global__ void kq_kernel(const float* __restrict__ x,
                          const float* __restrict__ w_k, const float* __restrict__ b_k,
                          const float* __restrict__ w_q, const float* __restrict__ b_q,
                          _Float16* __restrict__ kP16, _Float16* __restrict__ qP16) {
    int idx = blockIdx.x * 256 + threadIdx.x;     // b*16*16384 + c8*16384 + pos
    int pos = idx & 16383;
    int t = idx >> 14;
    int c8 = t & 15, b = t >> 4;
    const float* xb = x + (size_t)b * 128 * 16384 + pos;
    const float* wk = w_k + c8 * 128;
    const float* wq = w_q + c8 * 128;
    float ak = 0.f, aq = 0.f;
    for (int c = 0; c < 128; ++c) {
        float xv = xb[(size_t)c * 16384];
        ak += wk[c] * xv;
        aq += wq[c] * xv;
    }
    ak += b_k[c8];
    aq += b_q[c8];
    int dpos = pos >> 10, hw = pos & 1023;
    size_t o = ((size_t)b * 1024 + hw) * 256 + c8 * 16 + dpos;
    kP16[o] = (_Float16)ak;
    qP16[o] = (_Float16)aq;
}

// ---------------- stage 2: x -> f16 transposed [b][pos][cin] ----------------
__global__ void xcast_kernel(const float* __restrict__ x, _Float16* __restrict__ xT16) {
    size_t idx = (size_t)blockIdx.x * 256 + threadIdx.x;   // (b,c,pos)
    int pos = (int)(idx & 16383);
    int t = (int)(idx >> 14);
    int c = t & 127, b = t >> 7;
    xT16[((size_t)b * 16384 + pos) * 128 + c] = (_Float16)x[idx];
}

// ---------------- stage 3: w_v -> f16 per-offset [off][cout][cin] ----------------
__global__ void wcast_kernel(const float* __restrict__ w_v, _Float16* __restrict__ Wf16) {
    int idx = blockIdx.x * 256 + threadIdx.x;
    if (idx >= 128 * 128 * 27) return;
    int off = idx % 27;
    int t = idx / 27;
    int i = t & 127, o = t >> 7;
    Wf16[(size_t)off * 16384 + o * 128 + i] = (_Float16)w_v[idx];
}

// ---------------- stage 4: 3x3x3 conv as 27 shifted 128x128 GEMMs (WMMA) ----------------
// Block = 8 waves sharing one 32-cout strip (mt); each wave owns one (b, d, h) row of
// 32 positions and computes a 32x32 tile. Per offset, the 32x128 f16 weight panel is
// staged to LDS once via global_load_async_to_lds_b128 (ASYNCcnt) and shared.
__global__ void conv_wmma_kernel(const _Float16* __restrict__ xT16,
                                 const _Float16* __restrict__ Wf16,
                                 const float* __restrict__ b_v,
                                 float* __restrict__ v, _Float16* __restrict__ vP16) {
    __shared__ _Float16 sW[32 * 128];                // 8 KiB weight panel
    int lane = threadIdx.x & 31;
    int warp = threadIdx.x >> 5;
    int mt = blockIdx.x >> 9;                        // 0..3 cout strip, block-uniform
    int wv = (blockIdx.x & 511) * 8 + warp;          // 0..4095
    int b = wv >> 9;                                 // 0..7
    int row = wv & 511;                              // (d,h)
    int d = row >> 5, h = row & 31;
    int n = lane & 15, g = lane >> 4;
    const _Float16* xb = xT16 + (size_t)b * 16384 * 128;
    unsigned ldsbase = (unsigned)(unsigned long long)(&sW[0]);
    f32x8 acc[2][2] = {};
    for (int off = 0; off < 27; ++off) {
        // ---- async stage: 4096 halves = 512 x 16B, 2 chunks per thread ----
        {
            const _Float16* src = Wf16 + (size_t)off * 16384 + mt * 4096;
            int chunk = threadIdx.x * 2;
            #pragma unroll
            for (int j = 0; j < 2; ++j) {
                const _Float16* gs = src + (size_t)(chunk + j) * 8;
                unsigned la = ldsbase + (unsigned)(chunk + j) * 16u;
                asm volatile("global_load_async_to_lds_b128 %0, %1, off"
                             :: "v"(la), "v"(gs) : "memory");
            }
            asm volatile("s_wait_asynccnt 0" ::: "memory");
        }
        __syncthreads();
        int kd = off / 9, kh = (off / 3) % 3, kw = off % 3;
        int dd = d + kd - 1, hh = h + kh - 1;
        if (((unsigned)dd < 16u) && ((unsigned)hh < 32u)) {   // wave-uniform
            int ww0 = n + kw - 1;
            int ww1 = ww0 + 16;
            bool v0ok = (unsigned)ww0 < 32u;
            bool v1ok = (unsigned)ww1 < 32u;
            const _Float16* xrow = xb + (size_t)((dd << 10) + (hh << 5)) * 128;
            #pragma unroll
            for (int ksb = 0; ksb < 4; ++ksb) {
                int k0 = ksb * 32;
                // A frags from LDS (ds_load_b128 pairs)
                half16 a0, a1;
                {
                    const _Float16* p0 = sW + (size_t)n * 128 + k0 + g * 8;
                    H16u u; u.h[0] = *(const half8*)p0; u.h[1] = *(const half8*)(p0 + 16); a0 = u.v;
                    const _Float16* p1 = sW + (size_t)(16 + n) * 128 + k0 + g * 8;
                    H16u w; w.h[0] = *(const half8*)p1; w.h[1] = *(const half8*)(p1 + 16); a1 = w.v;
                }
                half16 bf0 = {}, bf1 = {};
                if (v0ok) {
                    const _Float16* p = xrow + (size_t)ww0 * 128 + k0 + g * 8;
                    H16u u; u.h[0] = *(const half8*)p; u.h[1] = *(const half8*)(p + 16); bf0 = u.v;
                }
                if (v1ok) {
                    const _Float16* p = xrow + (size_t)ww1 * 128 + k0 + g * 8;
                    H16u u; u.h[0] = *(const half8*)p; u.h[1] = *(const half8*)(p + 16); bf1 = u.v;
                }
                acc[0][0] = wmma16(a0, bf0, acc[0][0]);
                acc[0][1] = wmma16(a0, bf1, acc[0][1]);
                acc[1][0] = wmma16(a1, bf0, acc[1][0]);
                acc[1][1] = wmma16(a1, bf1, acc[1][1]);
            }
        }
        __syncthreads();
    }
    // epilogue: 32 couts x 32 positions
    int hwbase = h * 32;
    #pragma unroll
    for (int ti = 0; ti < 2; ++ti)
        #pragma unroll
        for (int tj = 0; tj < 2; ++tj)
            #pragma unroll
            for (int rr = 0; rr < 8; ++rr) {
                int cout = mt * 32 + ti * 16 + rr + 8 * g;
                int w = tj * 16 + n;
                int pos = (d << 10) + hwbase + w;
                float val = acc[ti][tj][rr] + b_v[cout];
                v[((size_t)b * 128 + cout) * 16384 + pos] = val;
                vP16[((size_t)b * 2048 + cout * 16 + d) * 1024 + hwbase + w] = (_Float16)val;
            }
}

// ---------------- stage 5: cor = K^T Q (1024x1024, K=256) per batch (WMMA, 32x32/wave) ----------------
__global__ void cor_wmma_kernel(const _Float16* __restrict__ kP16,
                                const _Float16* __restrict__ qP16,
                                float* __restrict__ cor) {
    int lane = threadIdx.x & 31;
    int wid = blockIdx.x * 8 + (threadIdx.x >> 5);   // 8192 waves
    int b = wid >> 10;                               // 1024 tiles per batch
    int r = wid & 1023;
    int mt = r >> 5, nt = r & 31;                    // 32x32 tiles
    const _Float16* A  = kP16 + (size_t)b * 1024 * 256;   // [s][f] row-major
    const _Float16* Bm = qP16 + (size_t)b * 1024 * 256;   // [t][f] = col-major over K
    f32x8 acc[2][2] = {};
    #pragma unroll
    for (int ks = 0; ks < 8; ++ks) {
        half16 a0 = load_a_frag(A,  mt * 32,      256, ks * 32, lane);
        half16 a1 = load_a_frag(A,  mt * 32 + 16, 256, ks * 32, lane);
        half16 b0 = load_b_frag(Bm, nt * 32,      256, ks * 32, lane);
        half16 b1 = load_b_frag(Bm, nt * 32 + 16, 256, ks * 32, lane);
        acc[0][0] = wmma16(a0, b0, acc[0][0]);
        acc[0][1] = wmma16(a0, b1, acc[0][1]);
        acc[1][0] = wmma16(a1, b0, acc[1][0]);
        acc[1][1] = wmma16(a1, b1, acc[1][1]);
    }
    int g = lane >> 4, nn = lane & 15;
    float* out = cor + ((size_t)b << 20);
    #pragma unroll
    for (int ti = 0; ti < 2; ++ti)
        #pragma unroll
        for (int tj = 0; tj < 2; ++tj)
            #pragma unroll
            for (int rr = 0; rr < 8; ++rr) {
                int rrow = mt * 32 + ti * 16 + rr + 8 * g;
                int col = nt * 32 + tj * 16 + nn;
                out[(size_t)rrow * 1024 + col] = acc[ti][tj][rr];
            }
}

// ---------------- stage 6: row softmax over t, output f16 A_p [b][j][t] ----------------
__global__ void softmax_p_kernel(const float* __restrict__ cor, _Float16* __restrict__ Ap16) {
    __shared__ float sm[256];
    int row = blockIdx.x;                            // b*1024 + s
    int tid = threadIdx.x;
    const float* rp = cor + (size_t)row * 1024;
    float v0 = rp[tid], v1 = rp[tid + 256], v2 = rp[tid + 512], v3 = rp[tid + 768];
    float mx = fmaxf(fmaxf(v0, v1), fmaxf(v2, v3));
    sm[tid] = mx; __syncthreads();
    for (int s = 128; s > 0; s >>= 1) { if (tid < s) sm[tid] = fmaxf(sm[tid], sm[tid + s]); __syncthreads(); }
    mx = sm[0]; __syncthreads();
    float e0 = __expf(v0 - mx), e1 = __expf(v1 - mx), e2 = __expf(v2 - mx), e3 = __expf(v3 - mx);
    sm[tid] = e0 + e1 + e2 + e3; __syncthreads();
    for (int s = 128; s > 0; s >>= 1) { if (tid < s) sm[tid] += sm[tid + s]; __syncthreads(); }
    float inv = 1.0f / sm[0];
    _Float16* op = Ap16 + (size_t)row * 1024;
    op[tid]       = (_Float16)(e0 * inv);
    op[tid + 256] = (_Float16)(e1 * inv);
    op[tid + 512] = (_Float16)(e2 * inv);
    op[tid + 768] = (_Float16)(e3 * inv);
}

// ---------------- stage 7: D-attention logits (16x16 per batch, K=16384) ----------------
__global__ void adlogits_kernel(const _Float16* __restrict__ kP16,
                                const _Float16* __restrict__ qP16,
                                float* __restrict__ Adlog) {
    __shared__ float sm[256];
    int blk = blockIdx.x;                            // b*256 + i*16 + j
    int b = blk >> 8, ij = blk & 255, i = ij >> 4, j = ij & 15;
    const _Float16* kb = kP16 + (size_t)b * 1024 * 256;
    const _Float16* qb = qP16 + (size_t)b * 1024 * 256;
    float acc = 0.f;
    for (int l = threadIdx.x; l < 16384; l += 256) {
        int hw = l >> 4, c8 = l & 15;
        int base = hw * 256 + c8 * 16;
        acc += (float)kb[base + i] * (float)qb[base + j];
    }
    sm[threadIdx.x] = acc; __syncthreads();
    for (int s = 128; s > 0; s >>= 1) { if (threadIdx.x < s) sm[threadIdx.x] += sm[threadIdx.x + s]; __syncthreads(); }
    if (threadIdx.x == 0) Adlog[blk] = sm[0];
}

__global__ void dsoftmax_kernel(const float* __restrict__ Adlog, float* __restrict__ Ad) {
    __shared__ float sm[256];
    int b = blockIdx.x, tid = threadIdx.x, i = tid >> 4;
    float val = Adlog[b * 256 + tid];
    sm[tid] = val; __syncthreads();
    float mx = -1e30f;
    for (int j2 = 0; j2 < 16; ++j2) mx = fmaxf(mx, sm[i * 16 + j2]);
    float sum = 0.f;
    for (int j2 = 0; j2 < 16; ++j2) sum += __expf(sm[i * 16 + j2] - mx);
    Ad[b * 256 + tid] = __expf(val - mx) / sum;
}

// ---------------- stage 8: Patt = vf @ A_p^T (2048x1024, K=1024) per batch (WMMA, 32x32/wave) ----------------
__global__ void patt_wmma_kernel(const _Float16* __restrict__ vP16,
                                 const _Float16* __restrict__ Ap16,
                                 float* __restrict__ Patt) {
    int lane = threadIdx.x & 31;
    int wid = blockIdx.x * 8 + (threadIdx.x >> 5);   // 16384 waves
    int b = wid >> 11;                               // 2048 tiles per batch
    int r = wid & 2047;
    int mt = r >> 5, nt = r & 31;                    // M: 64 strips, N: 32 strips
    const _Float16* A  = vP16 + (size_t)b * 2048 * 1024;  // [m][t] row-major
    const _Float16* Bm = Ap16 + ((size_t)b << 20);        // [j][t] = col-major over K=t
    int nn = lane & 15;
    f32x8 acc[2][2] = {};
    for (int ks = 0; ks < 32; ++ks) {
        half16 a0 = load_a_frag(A,  mt * 32,      1024, ks * 32, lane);
        half16 a1 = load_a_frag(A,  mt * 32 + 16, 1024, ks * 32, lane);
        half16 b0 = load_b_frag(Bm, nt * 32,      1024, ks * 32, lane);
        half16 b1 = load_b_frag(Bm, nt * 32 + 16, 1024, ks * 32, lane);
        if (ks < 31) {   // hint next K panel toward the caches (global_prefetch_b8)
            __builtin_prefetch(A  + (size_t)(mt * 32 + nn) * 1024 + (ks + 1) * 32, 0, 0);
            __builtin_prefetch(Bm + (size_t)(nt * 32 + nn) * 1024 + (ks + 1) * 32, 0, 0);
        }
        acc[0][0] = wmma16(a0, b0, acc[0][0]);
        acc[0][1] = wmma16(a0, b1, acc[0][1]);
        acc[1][0] = wmma16(a1, b0, acc[1][0]);
        acc[1][1] = wmma16(a1, b1, acc[1][1]);
    }
    int g = lane >> 4;
    #pragma unroll
    for (int ti = 0; ti < 2; ++ti)
        #pragma unroll
        for (int tj = 0; tj < 2; ++tj)
            #pragma unroll
            for (int rr = 0; rr < 8; ++rr) {
                int rrow = mt * 32 + ti * 16 + rr + 8 * g;
                int col = nt * 32 + tj * 16 + nn;
                Patt[((size_t)b * 2048 + rrow) * 1024 + col] = acc[ti][tj][rr];
            }
}

// ---------------- stage 9: out = gamma*(Patt + Datt) + x ----------------
__global__ void final_kernel(const float* __restrict__ x, const float* __restrict__ gamma,
                             const float* __restrict__ Patt, const float* __restrict__ v,
                             const float* __restrict__ Ad, float* __restrict__ out) {
    size_t idx = (size_t)blockIdx.x * 256 + threadIdx.x;  // (b,c,d,hw) flat = x layout
    int hw = (int)(idx & 1023);
    int t = (int)(idx >> 10);
    int d = t & 15;
    int bc = t >> 4;                                      // b*128 + c
    int c = bc & 127, b = bc >> 7;
    int m = c * 16 + d;
    float patt = Patt[((size_t)b * 2048 + m) * 1024 + hw];
    const float* vb = v + (size_t)bc * 16384;
    const float* ad = Ad + b * 256 + d * 16;
    float datt = 0.f;
    for (int d2 = 0; d2 < 16; ++d2) datt += ad[d2] * vb[(d2 << 10) + hw];
    out[idx] = gamma[0] * (patt + datt) + x[idx];
}

// ---------------- host launch ----------------
extern "C" void kernel_launch(void* const* d_in, const int* in_sizes, int n_in,
                              void* d_out, int out_size, void* d_ws, size_t ws_size,
                              hipStream_t stream) {
    const float* x     = (const float*)d_in[0];
    const float* gamma = (const float*)d_in[1];
    const float* w_k   = (const float*)d_in[2];
    const float* b_k   = (const float*)d_in[3];
    const float* w_q   = (const float*)d_in[4];
    const float* b_q   = (const float*)d_in[5];
    const float* w_v   = (const float*)d_in[6];
    const float* b_v   = (const float*)d_in[7];
    float* out = (float*)d_out;

    constexpr size_t MiB = 1ull << 20;
    char* ws = (char*)d_ws;
    _Float16* kP16  = (_Float16*)(ws + 0 * MiB);    //  4 MiB: [b][hw][f=256]
    _Float16* qP16  = (_Float16*)(ws + 4 * MiB);    //  4 MiB
    float*    cor   = (float*)   (ws + 8 * MiB);    // 32 MiB: [b][s][t]
    _Float16* Ap16  = (_Float16*)(ws + 40 * MiB);   // 16 MiB: [b][j][t]
    _Float16* xT16  = (_Float16*)(ws + 56 * MiB);   // 32 MiB: [b][pos][cin]
    _Float16* Wf16  = (_Float16*)(ws + 88 * MiB);   //  1 MiB: [off][cout][cin]
    float*    vbuf  = (float*)   (ws + 89 * MiB);   // 64 MiB: [b][c][pos]
    _Float16* vP16  = (_Float16*)(ws + 153 * MiB);  // 32 MiB: [b][c*16+d][hw]
    float*    Patt  = (float*)   (ws + 185 * MiB);  // 64 MiB: [b][m][j]
    float*    Adlog = (float*)   (ws + 249 * MiB);  //  8 KiB
    float*    Ad    = (float*)   (ws + 249 * MiB + 65536);

    kq_kernel<<<8192, 256, 0, stream>>>(x, w_k, b_k, w_q, b_q, kP16, qP16);
    xcast_kernel<<<65536, 256, 0, stream>>>(x, xT16);
    wcast_kernel<<<1728, 256, 0, stream>>>(w_v, Wf16);
    conv_wmma_kernel<<<2048, 256, 0, stream>>>(xT16, Wf16, b_v, vbuf, vP16);
    cor_wmma_kernel<<<1024, 256, 0, stream>>>(kP16, qP16, cor);
    softmax_p_kernel<<<8192, 256, 0, stream>>>(cor, Ap16);
    adlogits_kernel<<<2048, 256, 0, stream>>>(kP16, qP16, Adlog);
    dsoftmax_kernel<<<8, 256, 0, stream>>>(Adlog, Ad);
    patt_wmma_kernel<<<2048, 256, 0, stream>>>(vP16, Ap16, Patt);
    final_kernel<<<65536, 256, 0, stream>>>(x, gamma, Patt, vbuf, Ad, out);
}
